// LinearTransformerLayer_85341000172337
// MI455X (gfx1250) — matrix operand order repaired and checked by reading, not compiled, not run on hardware
//
#include <hip/hip_runtime.h>
#include <cstdint>
#include <cstddef>

// Problem constants (match reference)
#define B_   4
#define S_   4096
#define E_   1024
#define H_   16
#define D_   64
#define FF_  4096
#define MTOK (B_*S_)   // 16384 token rows

typedef __bf16 bf16;
typedef bf16  v8bf  __attribute__((ext_vector_type(8)));
typedef bf16  v16bf __attribute__((ext_vector_type(16)));
typedef float v8f   __attribute__((ext_vector_type(8)));
typedef unsigned int u32x4 __attribute__((ext_vector_type(4)));
typedef int i32x4 __attribute__((ext_vector_type(4)));
typedef int i32x8 __attribute__((ext_vector_type(8)));

static __device__ __forceinline__ v8f wmma_bf16(v16bf a, v16bf b, v8f c) {
  // D = A(16x32 bf16) x B(32x16 bf16) + C(16x16 f32)
  return __builtin_amdgcn_wmma_f32_16x16x32_bf16(false, a, false, b, (short)0, c, false, false);
}

static __device__ __forceinline__ float elu1(float x) {
  return x > 0.f ? x + 1.f : __expf(x);   // elu(x)+1
}

// Async global->LDS copy of 16 bytes per lane (ASYNCcnt tracked, bypasses VGPRs)
static __device__ __forceinline__ void async_copy_b128(unsigned lds_addr,
                                                       unsigned long long gaddr) {
  asm volatile("global_load_async_to_lds_b128 %0, %1, off"
               :: "v"(lds_addr), "v"(gaddr) : "memory");
}
static __device__ __forceinline__ void wait_async0() {
  asm volatile("s_wait_asynccnt 0" ::: "memory");
}

// TDM: 1-D contiguous tile (nelems bf16) -> LDS with hardware padding
// (pad_amount DWORDs inserted every 2^pad_interval 8-byte units).
static __device__ __forceinline__ void tdm_load_1d(unsigned lds_addr,
                                                   unsigned long long gaddr,
                                                   unsigned nelems,
                                                   unsigned pad_interval_code,
                                                   unsigned pad_amount_code) {
  u32x4 g0;
  g0[0] = 1u;                                            // count=1, user descriptor
  g0[1] = lds_addr;                                      // lds_addr[31:0]
  g0[2] = (unsigned)(gaddr & 0xFFFFFFFFull);             // global_addr[31:0]
  g0[3] = (unsigned)((gaddr >> 32) & 0x1FFFFFFull)       // global_addr[56:32]
          | (2u << 30);                                  // type = 2 ("image")
  i32x8 g1;
  g1[0] = (int)((1u << 16)                               // data_size = 2 bytes
              | (1u << 20)                               // pad_enable
              | (pad_interval_code << 22)
              | (pad_amount_code << 25));
  g1[1] = (int)((nelems & 0xFFFFu) << 16);               // tensor_dim0[15:0] @ bit48
  g1[2] = (int)(((nelems >> 16) & 0xFFFFu)               // tensor_dim0[31:16]
              | (1u << 16));                             // tensor_dim1 = 1 (lo)
  g1[3] = (int)((nelems & 0xFFFFu) << 16);               // tile_dim0 @ bit112
  g1[4] = (int)1;                                        // tile_dim1=1, tile_dim2=0
  g1[5] = (int)nelems;                                   // tensor_dim0_stride lo
  g1[6] = 0;
  g1[7] = 0;
  i32x4 z4 = {0, 0, 0, 0};
#if __clang_major__ >= 23
  i32x8 z8 = {0, 0, 0, 0, 0, 0, 0, 0};
  __builtin_amdgcn_tensor_load_to_lds(g0, g1, z4, z4, z8, 0);
#else
  __builtin_amdgcn_tensor_load_to_lds(g0, g1, z4, z4, 0);
#endif
}

// ---------------------------------------------------------------------------
// fp32 -> bf16 elementwise convert
// ---------------------------------------------------------------------------
__global__ __launch_bounds__(256)
void f32_to_bf16_kernel(const float* __restrict__ in, bf16* __restrict__ out, int n) {
  int i = (blockIdx.x * 256 + threadIdx.x) * 4;
  if (i + 3 < n) {
    float4 v = *(const float4*)(in + i);
    out[i + 0] = (bf16)v.x;
    out[i + 1] = (bf16)v.y;
    out[i + 2] = (bf16)v.z;
    out[i + 3] = (bf16)v.w;
  } else {
    for (int j = i; j < n; ++j) out[j] = (bf16)in[j];
  }
}

// ---------------------------------------------------------------------------
// Generic bf16 WMMA GEMM: C[M,N] = A[M,K] * B[K,N], per-element epilogue.
// Block tile 128x128, K-step 32, 256 threads = 8 waves.
// Double-buffered LDS; A staged via async global->LDS (overlaps with WMMA).
// Requires M%128==0, N%128==0, K%32==0.
// ---------------------------------------------------------------------------
template <typename Epi>
__global__ __launch_bounds__(256)
void gemm_bf16_wmma_kernel(const bf16* __restrict__ A, const bf16* __restrict__ Bm,
                           int M, int N, int K, Epi epi) {
  __shared__ bf16 As[2][128][40];   // [buf][m][k]  padded rows (16B aligned)
  __shared__ bf16 Bs[2][128][40];   // [buf][n][k]  B stored transposed

  const int tid  = threadIdx.x;
  const int lane = tid & 31;
  const int wave = tid >> 5;
  const int wm   = wave >> 1;          // 0..3
  const int wn   = wave & 1;           // 0..1
  const int m0   = blockIdx.y * 128;
  const int n0   = blockIdx.x * 128;
  const int hh   = lane >> 4;
  const int l15  = lane & 15;

  // staging coordinates
  const int rA  = tid >> 1;            // A row 0..127
  const int kcA = (tid & 1) * 16;      // A k-col 0/16
  const int kkB = tid >> 3;            // B k-row 0..31
  const int cB  = (tid & 7) * 16;      // B n-col 0..112

  v8f acc[2][4];
#pragma unroll
  for (int mt = 0; mt < 2; ++mt)
#pragma unroll
    for (int nt = 0; nt < 4; ++nt)
#pragma unroll
      for (int i = 0; i < 8; ++i) acc[mt][nt][i] = 0.f;

  auto issue_A = [&](int k0, int buf) {
    unsigned long long ga =
        (unsigned long long)(uintptr_t)(A + (size_t)(m0 + rA) * K + (k0 + kcA));
    unsigned la = (unsigned)(uintptr_t)&As[buf][rA][kcA];
    async_copy_b128(la, ga);
    async_copy_b128(la + 16, ga + 16);
  };
  auto load_B = [&](int k0, v8bf& b0, v8bf& b1) {
    const bf16* gb = Bm + (size_t)(k0 + kkB) * N + (n0 + cB);
    b0 = *(const v8bf*)gb;
    b1 = *(const v8bf*)(gb + 8);
  };
  auto store_B = [&](int buf, v8bf b0, v8bf b1) {
#pragma unroll
    for (int i = 0; i < 8; ++i) Bs[buf][cB + i][kkB] = b0[i];
#pragma unroll
    for (int i = 0; i < 8; ++i) Bs[buf][cB + 8 + i][kkB] = b1[i];
  };

  // prologue: stage tile 0
  v8bf b0, b1;
  issue_A(0, 0);
  load_B(0, b0, b1);
  store_B(0, b0, b1);
  wait_async0();
  __syncthreads();

  const int T = K / 32;
  for (int t = 0; t < T; ++t) {
    const int cur = t & 1, nxt = cur ^ 1;
    const bool more = (t + 1 < T);
    if (more) {                      // overlap next tile fill with compute
      issue_A((t + 1) * 32, nxt);
      load_B((t + 1) * 32, b0, b1);
    }

    v16bf afr[2], bfr[4];
#pragma unroll
    for (int mt = 0; mt < 2; ++mt) {
      int row = wm * 32 + mt * 16 + l15;
      const bf16* p0 = &As[cur][row][16 * hh];
      const bf16* p1 = &As[cur][row][16 * hh + 8];
#pragma unroll
      for (int i = 0; i < 8; ++i) { afr[mt][i] = p0[i]; afr[mt][8 + i] = p1[i]; }
    }
#pragma unroll
    for (int nt = 0; nt < 4; ++nt) {
      int coln = wn * 64 + nt * 16 + l15;
      const bf16* p = &Bs[cur][coln][16 * hh];
#pragma unroll
      for (int i = 0; i < 16; ++i) bfr[nt][i] = p[i];
    }
#pragma unroll
    for (int mt = 0; mt < 2; ++mt)
#pragma unroll
      for (int nt = 0; nt < 4; ++nt)
        acc[mt][nt] = wmma_bf16(afr[mt], bfr[nt], acc[mt][nt]);

    if (more) {
      store_B(nxt, b0, b1);
      wait_async0();                 // ASYNCcnt not covered by barrier
      __syncthreads();
    }
  }

  // epilogue: f32 C layout: vgpr v -> M = v + 8*half, N = lane&15
#pragma unroll
  for (int mt = 0; mt < 2; ++mt)
#pragma unroll
    for (int nt = 0; nt < 4; ++nt) {
      int col = n0 + wn * 64 + nt * 16 + l15;
#pragma unroll
      for (int v = 0; v < 8; ++v) {
        int row = m0 + wm * 32 + mt * 16 + v + 8 * hh;
        epi(row, col, acc[mt][nt][v]);
      }
    }
}

// ---------------------------------------------------------------------------
// Epilogue functors
// ---------------------------------------------------------------------------
struct EpiQKV {
  const float* bqkv;     // [3E]
  const float* mask;     // [B,S]
  const float* z_prev;   // [B,H,D,S]
  float* z_out;          // [B,H,D,S]
  bf16 *qb, *kb, *vb;    // [B,H,S,D]
  __device__ void operator()(int row, int col, float c) const {
    c += bqkv[col];
    int part = col >> 10;        // 0=q 1=k 2=v
    int e = col & (E_ - 1);
    int h = e >> 6, d = e & 63;
    int b = row >> 12, s = row & (S_ - 1);
    size_t hsidx = ((((size_t)b * H_ + h) * S_ + s) << 6) + d;
    if (part == 0) {
      qb[hsidx] = (bf16)elu1(c);
    } else if (part == 1) {
      float m = mask[(size_t)b * S_ + s];
      float kv = elu1(c) * m;
      kb[hsidx] = (bf16)kv;
      size_t zidx = (((size_t)b * H_ + h) * D_ + d) * (size_t)S_ + s;
      z_out[zidx] = z_prev[zidx] + kv;
    } else {
      float m = mask[(size_t)b * S_ + s];
      vb[hsidx] = (bf16)(c * m);
    }
  }
};

struct EpiBiasResid {   // out[row,col] = resid[row,col] + c + bias[col]
  const float* bias;
  const float* resid;
  float* out;
  int ld;
  __device__ void operator()(int row, int col, float c) const {
    size_t i = (size_t)row * ld + col;
    out[i] = resid[i] + c + bias[col];
  }
};

struct EpiReluBf {      // out[row,col] = bf16(max(c + bias, 0))
  const float* bias;
  bf16* out;
  int ld;
  __device__ void operator()(int row, int col, float c) const {
    c += bias[col];
    out[(size_t)row * ld + col] = (bf16)(c > 0.f ? c : 0.f);
  }
};

// ---------------------------------------------------------------------------
// S_t = S_prev + k^T v  per (b,h): 64x64 out, K=S=4096. 16 waves, 1 tile each.
// ---------------------------------------------------------------------------
__global__ __launch_bounds__(512)
void kv_state_kernel(const bf16* __restrict__ kb, const bf16* __restrict__ vb,
                     const float* __restrict__ S_prev, float* __restrict__ S_out) {
  const int bh = blockIdx.x;                 // 0..B*H-1
  __shared__ bf16 Ks[64][40];                // [d][s']  (k transposed)
  __shared__ bf16 Vs[64][40];                // [m][s']  (v transposed)
  const int tid  = threadIdx.x;
  const int lane = tid & 31;
  const int wave = tid >> 5;                 // 0..15
  const int hh   = lane >> 4;
  const int l15  = lane & 15;
  const int td   = wave >> 2;                // d-tile 0..3
  const int tm   = wave & 3;                 // m-tile 0..3

  v8f acc;
#pragma unroll
  for (int i = 0; i < 8; ++i) acc[i] = 0.f;

  const bf16* kbase = kb + (size_t)bh * S_ * D_;
  const bf16* vbase = vb + (size_t)bh * S_ * D_;

  const int si = tid >> 4;                   // 0..31
  const int dc = (tid & 15) * 4;             // 0..60

  for (int s0 = 0; s0 < S_; s0 += 32) {
    const bf16* gk = kbase + (size_t)(s0 + si) * D_ + dc;
    const bf16* gv = vbase + (size_t)(s0 + si) * D_ + dc;
#pragma unroll
    for (int j = 0; j < 4; ++j) { Ks[dc + j][si] = gk[j]; Vs[dc + j][si] = gv[j]; }
    __syncthreads();

    v16bf a, b;
    const bf16* p0 = &Ks[td * 16 + l15][16 * hh];
    const bf16* p1 = p0 + 8;
#pragma unroll
    for (int i = 0; i < 8; ++i) { a[i] = p0[i]; a[8 + i] = p1[i]; }
    const bf16* pb = &Vs[tm * 16 + l15][16 * hh];
#pragma unroll
    for (int i = 0; i < 16; ++i) b[i] = pb[i];

    acc = wmma_bf16(a, b, acc);
    __syncthreads();
  }

  const int col = tm * 16 + l15;
#pragma unroll
  for (int v = 0; v < 8; ++v) {
    int drow = td * 16 + v + 8 * hh;
    size_t idx = (size_t)bh * (D_ * D_) + (size_t)drow * D_ + col;
    S_out[idx] = S_prev[idx] + acc[v];
  }
}

// ---------------------------------------------------------------------------
// zsum[b,h,d] = sum_s k[b,h,s,d] + sum_s z_prev[b,h,d,s]
// ---------------------------------------------------------------------------
__global__ __launch_bounds__(256)
void zsum_kernel(const bf16* __restrict__ kb, const float* __restrict__ z_prev,
                 float* __restrict__ zsum) {
  const int bh = blockIdx.x;
  __shared__ float red[256];
  const int d = threadIdx.x & 63, chunk = threadIdx.x >> 6;
  float acc = 0.f;
  const bf16* kbase = kb + (size_t)bh * S_ * D_;
  for (int s = chunk; s < S_; s += 4) acc += (float)kbase[(size_t)s * D_ + d];
  const float* zb = z_prev + ((size_t)bh * D_ + d) * S_;
  for (int s = chunk; s < S_; s += 4) acc += zb[s];
  red[threadIdx.x] = acc;
  __syncthreads();
  if (threadIdx.x < 64)
    zsum[(size_t)bh * D_ + threadIdx.x] =
        red[threadIdx.x] + red[threadIdx.x + 64] + red[threadIdx.x + 128] + red[threadIdx.x + 192];
}

// den[b,h,s] = dot(q[b,h,s,:], zsum[b,h,:])
__global__ __launch_bounds__(256)
void den_kernel(const bf16* __restrict__ qb, const float* __restrict__ zsum,
                float* __restrict__ den) {
  size_t i = (size_t)blockIdx.x * 256 + threadIdx.x;   // B*H*S total
  int bh = (int)(i >> 12);
  const bf16* q = qb + (i << 6);
  const float* z = zsum + (size_t)bh * D_;
  float acc = 0.f;
#pragma unroll
  for (int d = 0; d < D_; ++d) acc += (float)q[d] * z[d];
  den[i] = acc;
}

// ---------------------------------------------------------------------------
// attn[b,s,h*64+m] = (q @ S_t)[s,m] / (den[s]+eps)  per (b,h); M=4096 N=64 K=64
// Q tile (16 KB contiguous) staged by the Tensor Data Mover with HW padding:
// 4 DWORDs pad every 32 DWORDs -> LDS rows of 72 bf16 (bank-conflict free).
// ---------------------------------------------------------------------------
__global__ __launch_bounds__(256)
void attn_kernel(const bf16* __restrict__ qb, const float* __restrict__ St,
                 const float* __restrict__ den, bf16* __restrict__ attn) {
  const int bh = blockIdx.x;
  const int s0 = blockIdx.y * 128;
  const int b  = bh >> 4, h = bh & 15;
  __shared__ bf16 Qs[128][72];   // [s][d]  filled by TDM (padded)
  __shared__ bf16 Bs[64][72];    // [m][d]  (S_t transposed)

  const int tid  = threadIdx.x;
  const int lane = tid & 31;
  const int wave = tid >> 5;
  const int hh   = lane >> 4;
  const int l15  = lane & 15;

  // TDM: wave 0 pulls the whole 128x64 bf16 Q panel into LDS.
  if (wave == 0) {
    tdm_load_1d((unsigned)(uintptr_t)&Qs[0][0],
                (unsigned long long)(uintptr_t)(qb + ((size_t)bh * S_ + s0) * D_),
                128 * 64, /*pad_interval: 32 DWORDs*/ 4, /*pad: 4 DWORDs*/ 3);
    __builtin_amdgcn_s_wait_tensorcnt(0);
  }
  // load S_t 64x64 f32, convert bf16, store transposed
  {
    int d  = tid >> 2;
    int c0 = (tid & 3) * 16;
    const float* gs = St + (size_t)bh * (D_ * D_) + (size_t)d * D_ + c0;
#pragma unroll
    for (int i = 0; i < 16; ++i) Bs[c0 + i][d] = (bf16)gs[i];
  }
  __syncthreads();

  v8f acc[4];
#pragma unroll
  for (int nt = 0; nt < 4; ++nt)
#pragma unroll
    for (int i = 0; i < 8; ++i) acc[nt][i] = 0.f;

#pragma unroll
  for (int kk0 = 0; kk0 < 64; kk0 += 32) {
    v16bf a;
    const bf16* p0 = &Qs[wave * 16 + l15][kk0 + 16 * hh];
    const bf16* p1 = &Qs[wave * 16 + l15][kk0 + 16 * hh + 8];
#pragma unroll
    for (int i = 0; i < 8; ++i) { a[i] = p0[i]; a[8 + i] = p1[i]; }
#pragma unroll
    for (int nt = 0; nt < 4; ++nt) {
      v16bf bfr;
      const bf16* pb = &Bs[nt * 16 + l15][kk0 + 16 * hh];
#pragma unroll
      for (int i = 0; i < 16; ++i) bfr[i] = pb[i];
      acc[nt] = wmma_bf16(a, bfr, acc[nt]);
    }
  }

#pragma unroll
  for (int nt = 0; nt < 4; ++nt) {
    int col = nt * 16 + l15;
#pragma unroll
    for (int v = 0; v < 8; ++v) {
      int s = s0 + wave * 16 + v + 8 * hh;
      float dv = den[(size_t)bh * S_ + s] + 1e-6f;
      float av = acc[nt][v] / dv;
      attn[((size_t)b * S_ + s) * E_ + h * 64 + col] = (bf16)av;
    }
  }
}

// ---------------------------------------------------------------------------
// LayerNorm over E=1024 per row; optional bf16 copy. In-place safe.
// ---------------------------------------------------------------------------
__global__ __launch_bounds__(256)
void layernorm_kernel(const float* __restrict__ in, const float* __restrict__ g,
                      const float* __restrict__ beta, float* __restrict__ out_f32,
                      bf16* __restrict__ out_bf) {
  const int row = blockIdx.x;
  const int tid = threadIdx.x;
  const float* x = in + (size_t)row * E_;
  __shared__ float red[256];

  float v[4], s = 0.f, s2 = 0.f;
#pragma unroll
  for (int i = 0; i < 4; ++i) {
    v[i] = x[tid + i * 256];
    s += v[i];
    s2 += v[i] * v[i];
  }
  red[tid] = s; __syncthreads();
  for (int off = 128; off > 0; off >>= 1) { if (tid < off) red[tid] += red[tid + off]; __syncthreads(); }
  float mean = red[0] * (1.f / E_);
  __syncthreads();
  red[tid] = s2; __syncthreads();
  for (int off = 128; off > 0; off >>= 1) { if (tid < off) red[tid] += red[tid + off]; __syncthreads(); }
  float var = red[0] * (1.f / E_) - mean * mean;
  float inv = rsqrtf(var + 1e-5f);

#pragma unroll
  for (int i = 0; i < 4; ++i) {
    int c = tid + i * 256;
    float y = (v[i] - mean) * inv * g[c] + beta[c];
    out_f32[(size_t)row * E_ + c] = y;
    if (out_bf) out_bf[(size_t)row * E_ + c] = (bf16)y;
  }
}

// ---------------------------------------------------------------------------
// Host driver
// ---------------------------------------------------------------------------
extern "C" void kernel_launch(void* const* d_in, const int* in_sizes, int n_in,
                              void* d_out, int out_size, void* d_ws, size_t ws_size,
                              hipStream_t stream) {
  const float* x      = (const float*)d_in[0];
  const float* mask   = (const float*)d_in[1];
  const float* S_prev = (const float*)d_in[2];
  const float* z_prev = (const float*)d_in[3];
  const float* Wqkv   = (const float*)d_in[4];
  const float* bqkv   = (const float*)d_in[5];
  const float* Wout   = (const float*)d_in[6];
  const float* bout   = (const float*)d_in[7];
  const float* ln1_g  = (const float*)d_in[8];
  const float* ln1_b  = (const float*)d_in[9];
  const float* ln2_g  = (const float*)d_in[10];
  const float* ln2_b  = (const float*)d_in[11];
  const float* W1     = (const float*)d_in[12];
  const float* b1     = (const float*)d_in[13];
  const float* W2     = (const float*)d_in[14];
  const float* b2     = (const float*)d_in[15];

  float* out    = (float*)d_out;
  float* x2_out = out;                                     // [B,S,E]
  float* St_out = out + (size_t)MTOK * E_;                 // [B,H,D,D]
  float* z_out  = St_out + (size_t)B_ * H_ * D_ * D_;      // [B,H,D,S]

  char* ws = (char*)d_ws;
  size_t off = 0;
  auto alloc = [&](size_t bytes) -> void* {
    void* p = ws + off;
    off += (bytes + 255) & ~(size_t)255;
    return p;
  };

  bf16* Wqkv_bf = (bf16*)alloc((size_t)E_ * 3 * E_ * sizeof(bf16));
  bf16* Wout_bf = (bf16*)alloc((size_t)E_ * E_ * sizeof(bf16));
  bf16* W1_bf   = (bf16*)alloc((size_t)E_ * FF_ * sizeof(bf16));
  bf16* W2_bf   = (bf16*)alloc((size_t)FF_ * E_ * sizeof(bf16));
  bf16* x_bf    = (bf16*)alloc((size_t)MTOK * E_ * sizeof(bf16));
  // big region: phase A = q|k|v|attn (4x 33.5MB), phase B = FFN hidden (134MB)
  char* big     = (char*)alloc((size_t)MTOK * FF_ * sizeof(bf16));
  bf16* q_bf    = (bf16*)(big);
  bf16* k_bf    = q_bf + (size_t)MTOK * E_;
  bf16* v_bf    = k_bf + (size_t)MTOK * E_;
  bf16* attn_bf = v_bf + (size_t)MTOK * E_;
  bf16* h_bf    = (bf16*)(big);                            // FFN hidden, reuses region
  float* x1_f32 = (float*)alloc((size_t)MTOK * E_ * sizeof(float));
  bf16* x1_bf   = (bf16*)alloc((size_t)MTOK * E_ * sizeof(bf16));
  float* zsum   = (float*)alloc((size_t)B_ * H_ * D_ * sizeof(float));
  float* den    = (float*)alloc((size_t)B_ * H_ * S_ * sizeof(float));

  auto cvt = [&](const float* src, bf16* dst, int n) {
    int grid = (n + 1023) / 1024;
    f32_to_bf16_kernel<<<grid, 256, 0, stream>>>(src, dst, n);
  };

  // 1) precision conversion of streamed GEMM operands
  cvt(Wqkv, Wqkv_bf, E_ * 3 * E_);
  cvt(Wout, Wout_bf, E_ * E_);
  cvt(W1, W1_bf, E_ * FF_);
  cvt(W2, W2_bf, FF_ * E_);
  cvt(x, x_bf, MTOK * E_);

  // 2) qkv GEMM with fused elu+1 / mask / z_t epilogue
  {
    EpiQKV e{bqkv, mask, z_prev, z_out, q_bf, k_bf, v_bf};
    gemm_bf16_wmma_kernel<EpiQKV>
        <<<dim3(3 * E_ / 128, MTOK / 128), 256, 0, stream>>>(x_bf, Wqkv_bf, MTOK, 3 * E_, E_, e);
  }

  // 3) zsum, S_t, den
  zsum_kernel<<<B_ * H_, 256, 0, stream>>>(k_bf, z_prev, zsum);
  kv_state_kernel<<<B_ * H_, 512, 0, stream>>>(k_bf, v_bf, S_prev, St_out);
  den_kernel<<<(B_ * H_ * S_) / 256, 256, 0, stream>>>(q_bf, zsum, den);

  // 4) num GEMM + 1/(den+eps), scatter to [B,S,E]
  attn_kernel<<<dim3(B_ * H_, S_ / 128), 256, 0, stream>>>(q_bf, St_out, den, attn_bf);

  // 5) out-proj GEMM + x residual -> x1 pre-LN
  {
    EpiBiasResid e{bout, x, x1_f32, E_};
    gemm_bf16_wmma_kernel<EpiBiasResid>
        <<<dim3(E_ / 128, MTOK / 128), 256, 0, stream>>>(attn_bf, Wout_bf, MTOK, E_, E_, e);
  }
  // 6) LN1 -> x1 (f32, in place) + x1 bf16
  layernorm_kernel<<<MTOK, 256, 0, stream>>>(x1_f32, ln1_g, ln1_b, x1_f32, x1_bf);

  // 7) FFN up GEMM + relu (bf16 hidden, reuses q/k/v/attn region)
  {
    EpiReluBf e{b1, h_bf, FF_};
    gemm_bf16_wmma_kernel<EpiReluBf>
        <<<dim3(FF_ / 128, MTOK / 128), 256, 0, stream>>>(x1_bf, W1_bf, MTOK, FF_, E_, e);
  }
  // 8) FFN down GEMM + x1 residual -> x2 pre-LN (direct to d_out)
  {
    EpiBiasResid e{b2, x1_f32, x2_out, E_};
    gemm_bf16_wmma_kernel<EpiBiasResid>
        <<<dim3(E_ / 128, MTOK / 128), 256, 0, stream>>>(h_bf, W2_bf, MTOK, E_, FF_, e);
  }
  // 9) LN2 in place on d_out
  layernorm_kernel<<<MTOK, 256, 0, stream>>>(x2_out, ln2_g, ln2_b, x2_out, nullptr);

  (void)in_sizes; (void)n_in; (void)out_size; (void)ws_size;
}
